// V5_BiDafAdjust_38173669327235
// MI455X (gfx1250) — compile-verified, matrix operand order checked by reading, not run on hardware
//
#include <hip/hip_runtime.h>
#include <hip/hip_bf16.h>

typedef __bf16 bf16;
typedef __attribute__((ext_vector_type(16))) __bf16 v16bf;
typedef __attribute__((ext_vector_type(8)))  float  v8f;

#define CEIL(a, b) (((a) + (b) - 1) / (b))

// ---------------------------------------------------------------------------
// WMMA helpers (CDNA5 layouts, ISA 7.12.2)
// A operand (16x32 bf16): lane 0-15 -> row M=lane, K = {k0..k0+7, k0+16..k0+23}
//                         lane 16-31 -> row M=lane-16, K = {k0+8..k0+15, k0+24..k0+31}
// B operand is symmetric with columns of B (rows of B^T).
// D: vgpr r, lane l -> row = r + (l>=16 ? 8 : 0), col = l & 15.
// ---------------------------------------------------------------------------
__device__ inline v8f wmma_bf16(v16bf a, v16bf b, v8f c) {
  return __builtin_amdgcn_wmma_f32_16x16x32_bf16(false, a, false, b, (short)0, c,
                                                 false, false);
}

// rows contiguous along K (row-major MxK or NxK)
__device__ inline v16bf load_k32(const bf16* base, int ld, int row0, int k0, int lane) {
  int r  = row0 + (lane & 15);
  int kb = k0 + ((lane & 16) ? 8 : 0);
  const bf16* p = base + (long long)r * ld + kb;
  v16bf v;
#pragma unroll
  for (int i = 0; i < 8; ++i) v[i] = p[i];
#pragma unroll
  for (int i = 0; i < 8; ++i) v[i + 8] = p[16 + i];
  return v;
}

// B stored KxN row-major: gather column n over the K pattern
__device__ inline v16bf load_k32_colK(const bf16* base, int ld, int n0, int k0, int lane) {
  int n  = n0 + (lane & 15);
  int kb = k0 + ((lane & 16) ? 8 : 0);
  v16bf v;
#pragma unroll
  for (int i = 0; i < 8; ++i) v[i] = base[(long long)(kb + i) * ld + n];
#pragma unroll
  for (int i = 0; i < 8; ++i) v[i + 8] = base[(long long)(kb + 16 + i) * ld + n];
  return v;
}

__device__ inline float sigm(float x) { return 1.f / (1.f + __expf(-x)); }

// ---------------------------------------------------------------------------
// Register-blocked WMMA GEMMs: one wave computes a 32x32 macro-tile
// (2x2 WMMA tiles) -> each A/B fragment feeds 2 WMMAs.
// Loop-unswitched: full tiles take a branchless inner loop; only edge blocks
// (M or N not multiple of 32) take the guarded path. All guards are uniform
// per-wave so EXEC stays all-ones around every WMMA.
// gemm_nt: C[b] = A[b](MxK) * B[b](NxK)^T (+bias[n])   (K multiple of 32)
// gemm_nn: C[b] = A[b](MxK) * B[b](KxN)
// ---------------------------------------------------------------------------
__global__ __launch_bounds__(32)
void gemm_nt(const bf16* __restrict__ A, const bf16* __restrict__ B,
             const float* __restrict__ bias, void* Cout, int bf16_out,
             int M, int N, int K, int lda, int ldb, int ldc,
             long long sA, long long sB, long long sC) {
  int lane = threadIdx.x;
  int n0 = blockIdx.x * 32, m0 = blockIdx.y * 32, b = blockIdx.z;
  int nm = (m0 + 16 < M) ? 2 : 1;
  int nn = (n0 + 16 < N) ? 2 : 1;
  const bf16* Ab = A + (long long)b * sA;
  const bf16* Bb = B + (long long)b * sB;
  float bv0 = bias ? bias[n0 + (lane & 15)] : 0.f;
  float bv1 = (bias && nn == 2) ? bias[n0 + 16 + (lane & 15)] : 0.f;
  v8f acc[2][2];
#pragma unroll
  for (int r = 0; r < 8; ++r) {
    acc[0][0][r] = bv0; acc[1][0][r] = bv0;
    acc[0][1][r] = bv1; acc[1][1][r] = bv1;
  }
  if (nm == 2 && nn == 2) {
    // fast path: branchless inner loop
    for (int k = 0; k < K; k += 32) {
      v16bf a0 = load_k32(Ab, lda, m0, k, lane);
      v16bf a1 = load_k32(Ab, lda, m0 + 16, k, lane);
      v16bf b0 = load_k32(Bb, ldb, n0, k, lane);
      v16bf b1 = load_k32(Bb, ldb, n0 + 16, k, lane);
      acc[0][0] = wmma_bf16(a0, b0, acc[0][0]);
      acc[0][1] = wmma_bf16(a0, b1, acc[0][1]);
      acc[1][0] = wmma_bf16(a1, b0, acc[1][0]);
      acc[1][1] = wmma_bf16(a1, b1, acc[1][1]);
    }
  } else {
    for (int k = 0; k < K; k += 32) {
      v16bf a0 = load_k32(Ab, lda, m0, k, lane);
      v16bf b0 = load_k32(Bb, ldb, n0, k, lane);
      v16bf a1 = (nm == 2) ? load_k32(Ab, lda, m0 + 16, k, lane) : a0;
      v16bf b1 = (nn == 2) ? load_k32(Bb, ldb, n0 + 16, k, lane) : b0;
      acc[0][0] = wmma_bf16(a0, b0, acc[0][0]);
      if (nn == 2)            acc[0][1] = wmma_bf16(a0, b1, acc[0][1]);
      if (nm == 2)            acc[1][0] = wmma_bf16(a1, b0, acc[1][0]);
      if (nm == 2 && nn == 2) acc[1][1] = wmma_bf16(a1, b1, acc[1][1]);
    }
  }
  int colL = lane & 15;
  int rbL  = (lane & 16) ? 8 : 0;
  for (int mt = 0; mt < nm; ++mt)
    for (int nt = 0; nt < nn; ++nt) {
      int rb = m0 + mt * 16 + rbL;
      int col = n0 + nt * 16 + colL;
      if (bf16_out) {
        bf16* C = (bf16*)Cout + (long long)b * sC;
#pragma unroll
        for (int r = 0; r < 8; ++r) C[(long long)(rb + r) * ldc + col] = (bf16)acc[mt][nt][r];
      } else {
        float* C = (float*)Cout + (long long)b * sC;
#pragma unroll
        for (int r = 0; r < 8; ++r) C[(long long)(rb + r) * ldc + col] = acc[mt][nt][r];
      }
    }
}

__global__ __launch_bounds__(32)
void gemm_nn(const bf16* __restrict__ A, const bf16* __restrict__ B, void* Cout,
             int bf16_out, int M, int N, int K, int lda, int ldb, int ldc,
             long long sA, long long sB, long long sC) {
  int lane = threadIdx.x;
  int n0 = blockIdx.x * 32, m0 = blockIdx.y * 32, b = blockIdx.z;
  int nm = (m0 + 16 < M) ? 2 : 1;
  int nn = (n0 + 16 < N) ? 2 : 1;
  const bf16* Ab = A + (long long)b * sA;
  const bf16* Bb = B + (long long)b * sB;
  v8f acc[2][2];
#pragma unroll
  for (int r = 0; r < 8; ++r) {
    acc[0][0][r] = 0.f; acc[0][1][r] = 0.f;
    acc[1][0][r] = 0.f; acc[1][1][r] = 0.f;
  }
  if (nm == 2 && nn == 2) {
    for (int k = 0; k < K; k += 32) {
      v16bf a0 = load_k32(Ab, lda, m0, k, lane);
      v16bf a1 = load_k32(Ab, lda, m0 + 16, k, lane);
      v16bf b0 = load_k32_colK(Bb, ldb, n0, k, lane);
      v16bf b1 = load_k32_colK(Bb, ldb, n0 + 16, k, lane);
      acc[0][0] = wmma_bf16(a0, b0, acc[0][0]);
      acc[0][1] = wmma_bf16(a0, b1, acc[0][1]);
      acc[1][0] = wmma_bf16(a1, b0, acc[1][0]);
      acc[1][1] = wmma_bf16(a1, b1, acc[1][1]);
    }
  } else {
    for (int k = 0; k < K; k += 32) {
      v16bf a0 = load_k32(Ab, lda, m0, k, lane);
      v16bf b0 = load_k32_colK(Bb, ldb, n0, k, lane);
      v16bf a1 = (nm == 2) ? load_k32(Ab, lda, m0 + 16, k, lane) : a0;
      v16bf b1 = (nn == 2) ? load_k32_colK(Bb, ldb, n0 + 16, k, lane) : b0;
      acc[0][0] = wmma_bf16(a0, b0, acc[0][0]);
      if (nn == 2)            acc[0][1] = wmma_bf16(a0, b1, acc[0][1]);
      if (nm == 2)            acc[1][0] = wmma_bf16(a1, b0, acc[1][0]);
      if (nm == 2 && nn == 2) acc[1][1] = wmma_bf16(a1, b1, acc[1][1]);
    }
  }
  int colL = lane & 15;
  int rbL  = (lane & 16) ? 8 : 0;
  for (int mt = 0; mt < nm; ++mt)
    for (int nt = 0; nt < nn; ++nt) {
      int rb = m0 + mt * 16 + rbL;
      int col = n0 + nt * 16 + colL;
      if (bf16_out) {
        bf16* C = (bf16*)Cout + (long long)b * sC;
#pragma unroll
        for (int r = 0; r < 8; ++r) C[(long long)(rb + r) * ldc + col] = (bf16)acc[mt][nt][r];
      } else {
        float* C = (float*)Cout + (long long)b * sC;
#pragma unroll
        for (int r = 0; r < 8; ++r) C[(long long)(rb + r) * ldc + col] = acc[mt][nt][r];
      }
    }
}

// ---------------------------------------------------------------------------
// BiDAF sim tile kernel: sim = wdot * (c·q) + cw[row] + qw[col] + att_b
// ---------------------------------------------------------------------------
__global__ __launch_bounds__(32)
void sim_kernel(const bf16* __restrict__ Cb, const bf16* __restrict__ Qb,
                const float* __restrict__ cw, const float* __restrict__ qw,
                const float* __restrict__ attw, const float* __restrict__ attb,
                float* __restrict__ sim) {
  int lane = threadIdx.x;
  int n0 = blockIdx.x * 16, m0 = blockIdx.y * 16, b = blockIdx.z;
  const bf16* A  = Cb + (long long)b * 400 * 256;
  const bf16* Bm = Qb + (long long)b * 48 * 256;
  v8f acc;
#pragma unroll
  for (int r = 0; r < 8; ++r) acc[r] = 0.f;
  for (int k = 0; k < 256; k += 32) {
    v16bf a = load_k32(A, 256, m0, k, lane);
    v16bf q = load_k32(Bm, 256, n0, k, lane);
    acc = wmma_bf16(a, q, acc);
  }
  float wdot = attw[512], ab = attb[0];
  int col = n0 + (lane & 15);
  int rb  = m0 + ((lane & 16) ? 8 : 0);
#pragma unroll
  for (int r = 0; r < 8; ++r) {
    int row = rb + r;
    sim[((long long)b * 400 + row) * 48 + col] =
        wdot * acc[r] + cw[b * 400 + row] + qw[b * 48 + col] + ab;
  }
}

// ---------------------------------------------------------------------------
// Persistent LSTM recurrence: one block (8 waves), whole time loop.
// gates(32x512) = xw[t] + h(32x128) @ U(512x128)^T  via WMMA.
// Each wave owns cells [16w,16w+16): its 64x128 slice of U is hoisted into
// 128 VGPRs (16 v16bf fragments) ONCE, so the sequential critical path per
// step is only 8 LDS h-tile reads + 32 WMMAs + gate VALU. Cell state lives in
// registers (all 4 gates of a cell land in the same lane). h shared via 8KB
// LDS. Next step's xw lines are prefetched (global_prefetch_b8).
// ---------------------------------------------------------------------------
__global__ __launch_bounds__(256)
void lstm_recur(const float* __restrict__ xw, const bf16* __restrict__ U,
                float* __restrict__ out, int T, int reverse, int outOff) {
  __shared__ bf16 hsh[32 * 128];
  int tid = threadIdx.x;
  int w = tid >> 5, lane = tid & 31;
  for (int i = tid; i < 32 * 128; i += 256) hsh[i] = (bf16)0.f;
  float cst[2][8];
#pragma unroll
  for (int mt = 0; mt < 2; ++mt)
#pragma unroll
    for (int r = 0; r < 8; ++r) cst[mt][r] = 0.f;

  // hoist this wave's U slice into registers (16 fragments = 128 VGPRs)
  v16bf ureg[4][4];
#pragma unroll
  for (int kc = 0; kc < 4; ++kc)
#pragma unroll
    for (int gq = 0; gq < 4; ++gq)
      ureg[gq][kc] = load_k32(U, 128, gq * 128 + w * 16, kc * 32, lane);
  __syncthreads();

  int hi8  = (lane & 16) ? 8 : 0;
  int ncol = w * 16 + (lane & 15);   // cell index 0..127

  for (int step = 0; step < T; ++step) {
    int t = reverse ? (T - 1 - step) : step;
    v8f acc[2][4];
#pragma unroll
    for (int mt = 0; mt < 2; ++mt)
#pragma unroll
      for (int gq = 0; gq < 4; ++gq)
#pragma unroll
        for (int r = 0; r < 8; ++r) {
          int bb = mt * 16 + r + hi8;
          acc[mt][gq][r] = xw[((long long)bb * T + t) * 512 + gq * 128 + ncol];
        }
    // prefetch next step's xw cachelines
    if (step + 1 < T) {
      int tn = reverse ? (T - 2 - step) : (step + 1);
#pragma unroll
      for (int mt = 0; mt < 2; ++mt)
#pragma unroll
        for (int gq = 0; gq < 4; ++gq) {
          int bb = mt * 16 + hi8;
          __builtin_prefetch(&xw[((long long)bb * T + tn) * 512 + gq * 128 + ncol], 0, 1);
        }
    }
#pragma unroll
    for (int kc = 0; kc < 4; ++kc) {
      v16bf a0 = load_k32(hsh, 128, 0, kc * 32, lane);
      v16bf a1 = load_k32(hsh, 128, 16, kc * 32, lane);
#pragma unroll
      for (int gq = 0; gq < 4; ++gq) {
        acc[0][gq] = wmma_bf16(a0, ureg[gq][kc], acc[0][gq]);
        acc[1][gq] = wmma_bf16(a1, ureg[gq][kc], acc[1][gq]);
      }
    }
    __syncthreads();   // all reads of hsh done
#pragma unroll
    for (int mt = 0; mt < 2; ++mt)
#pragma unroll
      for (int r = 0; r < 8; ++r) {
        float iv = sigm(acc[mt][0][r]);
        float fv = sigm(acc[mt][1][r]);
        float gv = tanhf(acc[mt][2][r]);
        float ov = sigm(acc[mt][3][r]);
        float c = fv * cst[mt][r] + iv * gv;
        cst[mt][r] = c;
        float h = ov * tanhf(c);
        int bb = mt * 16 + r + hi8;
        hsh[bb * 128 + ncol] = (bf16)h;
        out[((long long)bb * T + t) * 256 + outOff + ncol] = h;
      }
    __syncthreads();
  }
}

// ---------------------------------------------------------------------------
// Utility kernels
// ---------------------------------------------------------------------------
__global__ void gather_emb(const int* __restrict__ tok, const float* __restrict__ embed,
                           bf16* __restrict__ dst, long long rows, int ldd) {
  long long i = (long long)blockIdx.x * blockDim.x + threadIdx.x;
  if (i >= rows * ldd) return;
  int c = (int)(i % ldd);
  long long r = i / ldd;
  dst[i] = (c < 300) ? (bf16)embed[(long long)tok[r] * 300 + c] : (bf16)0.f;
}

__global__ void cvt_bf16_pad(const float* __restrict__ src, bf16* __restrict__ dst,
                             long long rows, int cols, int lds, int ldd) {
  long long i = (long long)blockIdx.x * blockDim.x + threadIdx.x;
  if (i >= rows * ldd) return;
  int c = (int)(i % ldd);
  long long r = i / ldd;
  dst[i] = (c < cols) ? (bf16)src[r * lds + c] : (bf16)0.f;
}

__global__ void cvt_bf16_slice(const float* __restrict__ src, bf16* __restrict__ dst,
                               long long rows, int cols, int lds, int ldd) {
  long long i = (long long)blockIdx.x * blockDim.x + threadIdx.x;
  if (i >= rows * cols) return;
  int c = (int)(i % cols);
  long long r = i / cols;
  dst[r * ldd + c] = (bf16)src[r * lds + c];
}

__global__ void cvt_padrows(const float* __restrict__ src, bf16* __restrict__ dst,
                            int batch, int rows, int prow, int cols,
                            long long sSrc, long long sDst) {
  long long i = (long long)blockIdx.x * blockDim.x + threadIdx.x;
  long long n = (long long)batch * prow * cols;
  if (i >= n) return;
  int c = (int)(i % cols);
  long long rr = i / cols;
  int r = (int)(rr % prow);
  int b = (int)(rr / prow);
  float v = (r < rows) ? src[b * sSrc + (long long)r * cols + c] : 0.f;
  dst[b * sDst + (long long)r * cols + c] = (bf16)v;
}

__global__ void add_bias(const float* a, const float* b, float* out, int n) {
  int i = blockIdx.x * blockDim.x + threadIdx.x;
  if (i < n) out[i] = a[i] + b[i];
}

__global__ void rowdot(const float* __restrict__ X, const float* __restrict__ w,
                       float* __restrict__ out, long long rows, int D) {
  long long r = (long long)blockIdx.x * blockDim.x + threadIdx.x;
  if (r >= rows) return;
  const float* x = X + r * D;
  float s = 0.f;
  for (int k = 0; k < D; ++k) s += x[k] * w[k];
  out[r] = s;
}

__global__ __launch_bounds__(64)
void softmax_q(const float* __restrict__ sim, bf16* __restrict__ probs,
               float* __restrict__ rowmax) {
  __shared__ float red[64];
  long long row = blockIdx.x;
  int tid = threadIdx.x;
  const float* s = sim + row * 48;
  float v = (tid < 48) ? s[tid] : -3.4e38f;
  red[tid] = v; __syncthreads();
  for (int o = 32; o > 0; o >>= 1) { if (tid < o) red[tid] = fmaxf(red[tid], red[tid + o]); __syncthreads(); }
  float mx = red[0]; __syncthreads();
  float e = (tid < 48) ? __expf(v - mx) : 0.f;
  red[tid] = e; __syncthreads();
  for (int o = 32; o > 0; o >>= 1) { if (tid < o) red[tid] += red[tid + o]; __syncthreads(); }
  float sum = red[0];
  if (tid == 0) rowmax[row] = mx;
  probs[row * 64 + tid] = (tid < 48) ? (bf16)(e / sum) : (bf16)0.f;
}

__global__ __launch_bounds__(512)
void softmax_c(const float* __restrict__ rowmax, float* __restrict__ qcatt) {
  __shared__ float red[512];
  int b = blockIdx.x, tid = threadIdx.x;
  float v = (tid < 400) ? rowmax[b * 400 + tid] : -3.4e38f;
  red[tid] = v; __syncthreads();
  for (int o = 256; o > 0; o >>= 1) { if (tid < o) red[tid] = fmaxf(red[tid], red[tid + o]); __syncthreads(); }
  float mx = red[0]; __syncthreads();
  float e = (tid < 400) ? __expf(v - mx) : 0.f;
  red[tid] = e; __syncthreads();
  for (int o = 256; o > 0; o >>= 1) { if (tid < o) red[tid] += red[tid + o]; __syncthreads(); }
  float sum = red[0];
  if (tid < 400) qcatt[b * 400 + tid] = e / sum;
}

__global__ __launch_bounds__(512)
void softmax_row(const float* __restrict__ sc, bf16* __restrict__ probs,
                 int cols, int ldp) {
  __shared__ float red[512];
  long long row = blockIdx.x;
  int tid = threadIdx.x;
  const float* s = sc + row * cols;
  float v = (tid < cols) ? s[tid] : -3.4e38f;
  red[tid] = v; __syncthreads();
  for (int o = 256; o > 0; o >>= 1) { if (tid < o) red[tid] = fmaxf(red[tid], red[tid + o]); __syncthreads(); }
  float mx = red[0]; __syncthreads();
  float e = (tid < cols) ? __expf(v - mx) : 0.f;
  red[tid] = e; __syncthreads();
  for (int o = 256; o > 0; o >>= 1) { if (tid < o) red[tid] += red[tid + o]; __syncthreads(); }
  float sum = red[0];
  if (tid < ldp) probs[row * ldp + tid] = (tid < cols) ? (bf16)(e / sum) : (bf16)0.f;
}

__global__ __launch_bounds__(256)
void qc_emb_k(const float* __restrict__ qcatt, const float* __restrict__ cOut,
              float* __restrict__ qc) {
  int b = blockIdx.x, e = threadIdx.x;
  float s = 0.f;
  for (int c = 0; c < 400; ++c)
    s += qcatt[b * 400 + c] * cOut[((long long)b * 400 + c) * 256 + e];
  qc[b * 256 + e] = s;
}

__global__ void g_products(const float* __restrict__ cOut, const float* __restrict__ cq,
                           const float* __restrict__ qc, bf16* __restrict__ g) {
  long long i = (long long)blockIdx.x * blockDim.x + threadIdx.x;
  if (i >= 12800LL * 256) return;
  int e = (int)(i & 255);
  long long row = i >> 8;
  int b = (int)(row / 400);
  float c = cOut[i];
  g[row * 1536 + 512 + e] = (bf16)(c * cq[i]);
  g[row * 1536 + 768 + e] = (bf16)(c * qc[b * 256 + e]);
}

__global__ void finals(const float* __restrict__ m, const float* __restrict__ xend,
                       const float* __restrict__ sw, const float* __restrict__ sb,
                       const float* __restrict__ ew, const float* __restrict__ eb,
                       float* __restrict__ out) {
  long long i = (long long)blockIdx.x * blockDim.x + threadIdx.x;
  if (i >= 12800) return;
  int b = (int)(i / 400), t = (int)(i % 400);
  const float* mp = m + i * 256;
  const float* xp = xend + i * 256;
  float s = sb[0], e = eb[0];
  for (int k = 0; k < 256; ++k) {
    s += mp[k] * sw[k];
    e += mp[k] * ew[k] + xp[k] * ew[256 + k];
  }
  out[(long long)b * 800 + t] = s;
  out[(long long)b * 800 + 400 + t] = e;
}

// ---------------------------------------------------------------------------
// Host orchestration
// ---------------------------------------------------------------------------
extern "C" void kernel_launch(void* const* d_in, const int* in_sizes, int n_in,
                              void* d_out, int out_size, void* d_ws, size_t ws_size,
                              hipStream_t stream) {
  (void)in_sizes; (void)n_in; (void)out_size; (void)ws_size;
  const long long MQ = 32LL * 48, MC = 32LL * 400;

  char* wsb = (char*)d_ws;
  size_t off = 0;
  auto alloc = [&](size_t bytes) -> void* {
    void* p = wsb + off;
    off = (off + bytes + 255) & ~(size_t)255;
    return p;
  };

  const int* q_tok = (const int*)d_in[0];
  const int* c_tok = (const int*)d_in[1];
  const float* embed = (const float*)d_in[2];
  const float* attw = (const float*)d_in[43];
  const float* attb = (const float*)d_in[44];
  const float* sw = (const float*)d_in[45];
  const float* sb = (const float*)d_in[46];
  const float* ew = (const float*)d_in[47];
  const float* eb = (const float*)d_in[48];

  // activations
  bf16* qe = (bf16*)alloc(MQ * 320 * 2);
  bf16* ce = (bf16*)alloc(MC * 320 * 2);
  float* xwf = (float*)alloc(MC * 512 * 4);
  float* xwb = (float*)alloc(MC * 512 * 4);
  float* qo0 = (float*)alloc(MQ * 256 * 4);
  float* qo1 = (float*)alloc(MQ * 256 * 4);
  float* co0 = (float*)alloc(MC * 256 * 4);
  float* co1 = (float*)alloc(MC * 256 * 4);
  bf16* qi = (bf16*)alloc(MQ * 256 * 2);
  bf16* ci = (bf16*)alloc(MC * 256 * 2);

  // weight conversion (bf16, K zero-padded)
  auto cvtW = [&](const float* src, int rows, int cols, int ldd) -> bf16* {
    bf16* dst = (bf16*)alloc((size_t)rows * ldd * 2);
    long long n = (long long)rows * ldd;
    cvt_bf16_pad<<<CEIL(n, 256), 256, 0, stream>>>(src, dst, rows, cols, cols, ldd);
    return dst;
  };
  auto mkbias = [&](const float* b1, const float* b2) -> float* {
    float* d = (float*)alloc(512 * 4);
    add_bias<<<2, 256, 0, stream>>>(b1, b2, d, 512);
    return d;
  };
  struct LayerW { bf16 *Wf, *Uf, *Wb, *Ub; float *bf_, *bb_; int ldx; };
  auto mkLayer = [&](int base, int d, int dpad) -> LayerW {
    LayerW L;
    L.Wf = cvtW((const float*)d_in[base + 0], 512, d, dpad);
    L.Uf = cvtW((const float*)d_in[base + 1], 512, 128, 128);
    L.bf_ = mkbias((const float*)d_in[base + 2], (const float*)d_in[base + 3]);
    L.Wb = cvtW((const float*)d_in[base + 4], 512, d, dpad);
    L.Ub = cvtW((const float*)d_in[base + 5], 512, 128, 128);
    L.bb_ = mkbias((const float*)d_in[base + 6], (const float*)d_in[base + 7]);
    L.ldx = dpad;
    return L;
  };
  LayerW r0  = mkLayer(3, 300, 320);
  LayerW r1  = mkLayer(11, 256, 256);
  LayerW cq0 = mkLayer(19, 1536, 1536);
  LayerW cq1 = mkLayer(27, 256, 256);
  LayerW e0  = mkLayer(35, 256, 256);

  auto runLayer = [&](const bf16* x, const LayerW& L, int T, float* out) {
    int M = 32 * T;
    dim3 g(CEIL(512, 32), CEIL(M, 32), 1);
    gemm_nt<<<g, 32, 0, stream>>>(x, L.Wf, L.bf_, xwf, 0, M, 512, L.ldx,
                                  L.ldx, L.ldx, 512, 0, 0, 0);
    gemm_nt<<<g, 32, 0, stream>>>(x, L.Wb, L.bb_, xwb, 0, M, 512, L.ldx,
                                  L.ldx, L.ldx, 512, 0, 0, 0);
    lstm_recur<<<1, 256, 0, stream>>>(xwf, L.Uf, out, T, 0, 0);
    lstm_recur<<<1, 256, 0, stream>>>(xwb, L.Ub, out, T, 1, 128);
  };

  // embeddings
  gather_emb<<<CEIL(MQ * 320, 256), 256, 0, stream>>>(q_tok, embed, qe, MQ, 320);
  gather_emb<<<CEIL(MC * 320, 256), 256, 0, stream>>>(c_tok, embed, ce, MC, 320);

  // encoder biLSTMs (shared params for q and c)
  runLayer(qe, r0, 48, qo0);
  cvt_bf16_pad<<<CEIL(MQ * 256, 256), 256, 0, stream>>>(qo0, qi, MQ, 256, 256, 256);
  runLayer(qi, r1, 48, qo1);

  runLayer(ce, r0, 400, co0);
  cvt_bf16_pad<<<CEIL(MC * 256, 256), 256, 0, stream>>>(co0, ci, MC, 256, 256, 256);
  runLayer(ci, r1, 400, co1);

  // bf16 copies of final encoder outputs
  cvt_bf16_pad<<<CEIL(MC * 256, 256), 256, 0, stream>>>(co1, ci, MC, 256, 256, 256);
  cvt_bf16_pad<<<CEIL(MQ * 256, 256), 256, 0, stream>>>(qo1, qi, MQ, 256, 256, 256);

  // BiDAF attention
  float* cw = (float*)alloc(MC * 4);
  float* qw = (float*)alloc(MQ * 4);
  float* sim = (float*)alloc(MC * 48 * 4);
  float* rowmax = (float*)alloc(MC * 4);
  bf16* cqp = (bf16*)alloc(MC * 64 * 2);
  float* qcatt = (float*)alloc(MC * 4);
  float* qcv = (float*)alloc(32LL * 256 * 4);
  bf16* qK = (bf16*)alloc(32LL * 64 * 256 * 2);
  float* cq = (float*)alloc(MC * 256 * 4);
  bf16* gbf = (bf16*)alloc(MC * 1536 * 2);
  float* scores = (float*)alloc(MC * 400 * 4);
  bf16* probs = (bf16*)alloc(MC * 416 * 2);
  bf16* cK = (bf16*)alloc(32LL * 416 * 256 * 2);
  bf16* cqA = (bf16*)alloc(MC * 256 * 2);
  bf16* cqK = (bf16*)alloc(32LL * 416 * 256 * 2);

  rowdot<<<CEIL(MC, 256), 256, 0, stream>>>(co1, attw, cw, MC, 256);
  rowdot<<<CEIL(MQ, 256), 256, 0, stream>>>(qo1, attw + 256, qw, MQ, 256);
  sim_kernel<<<dim3(3, 25, 32), 32, 0, stream>>>(ci, qi, cw, qw, attw, attb, sim);
  softmax_q<<<(unsigned)MC, 64, 0, stream>>>(sim, cqp, rowmax);
  softmax_c<<<32, 512, 0, stream>>>(rowmax, qcatt);
  qc_emb_k<<<32, 256, 0, stream>>>(qcatt, co1, qcv);
  cvt_padrows<<<CEIL(32LL * 64 * 256, 256), 256, 0, stream>>>(qo1, qK, 32, 48, 64,
                                                              256, 48LL * 256, 64LL * 256);
  gemm_nn<<<dim3(CEIL(256, 32), CEIL(400, 32), 32), 32, 0, stream>>>(
      cqp, qK, cq, 0, 400, 256, 64, 64, 256, 256,
      400LL * 64, 64LL * 256, 400LL * 256);

  // g = [c | cq | c*cq | c*qc | self_att(c,c) | cross_att(c,cq)]  (bf16)
  cvt_bf16_slice<<<CEIL(MC * 256, 256), 256, 0, stream>>>(co1, gbf, MC, 256, 256, 1536);
  cvt_bf16_slice<<<CEIL(MC * 256, 256), 256, 0, stream>>>(cq, gbf + 256, MC, 256, 256, 1536);
  g_products<<<CEIL(MC * 256, 256), 256, 0, stream>>>(co1, cq, qcv, gbf);

  // self cross-feature
  gemm_nt<<<dim3(CEIL(400, 32), CEIL(400, 32), 32), 32, 0, stream>>>(
      ci, ci, nullptr, scores, 0, 400, 400, 256, 256, 256, 400,
      400LL * 256, 400LL * 256, 400LL * 400);
  softmax_row<<<(unsigned)MC, 512, 0, stream>>>(scores, probs, 400, 416);
  cvt_padrows<<<CEIL(32LL * 416 * 256, 256), 256, 0, stream>>>(co1, cK, 32, 400, 416,
                                                               256, 400LL * 256, 416LL * 256);
  gemm_nn<<<dim3(CEIL(256, 32), CEIL(400, 32), 32), 32, 0, stream>>>(
      probs, cK, (void*)(gbf + 1024), 1, 400, 256, 416, 416, 256, 1536,
      400LL * 416, 416LL * 256, 400LL * 1536);
  // cq cross-feature
  cvt_bf16_pad<<<CEIL(MC * 256, 256), 256, 0, stream>>>(cq, cqA, MC, 256, 256, 256);
  gemm_nt<<<dim3(CEIL(400, 32), CEIL(400, 32), 32), 32, 0, stream>>>(
      ci, cqA, nullptr, scores, 0, 400, 400, 256, 256, 256, 400,
      400LL * 256, 400LL * 256, 400LL * 400);
  softmax_row<<<(unsigned)MC, 512, 0, stream>>>(scores, probs, 400, 416);
  cvt_padrows<<<CEIL(32LL * 416 * 256, 256), 256, 0, stream>>>(cq, cqK, 32, 400, 416,
                                                               256, 400LL * 256, 416LL * 256);
  gemm_nn<<<dim3(CEIL(256, 32), CEIL(400, 32), 32), 32, 0, stream>>>(
      probs, cqK, (void*)(gbf + 1280), 1, 400, 256, 416, 416, 256, 1536,
      400LL * 416, 416LL * 256, 400LL * 1536);

  // modeling biLSTM (2 layers) over g -> m
  runLayer(gbf, cq0, 400, co0);
  cvt_bf16_pad<<<CEIL(MC * 256, 256), 256, 0, stream>>>(co0, ci, MC, 256, 256, 256);
  runLayer(ci, cq1, 400, co1);            // m = co1

  // end biLSTM over m -> x_end
  cvt_bf16_pad<<<CEIL(MC * 256, 256), 256, 0, stream>>>(co1, ci, MC, 256, 256, 256);
  runLayer(ci, e0, 400, co0);             // x_end = co0

  finals<<<CEIL(12800, 256), 256, 0, stream>>>(co1, co0, sw, sb, ew, eb, (float*)d_out);
}
